// SwitchFFN_13984413516052
// MI455X (gfx1250) — matrix operand order repaired and checked by reading, not compiled, not run on hardware
//
#include <hip/hip_runtime.h>

typedef __attribute__((ext_vector_type(16))) __bf16        v16bf;
typedef __attribute__((ext_vector_type(8)))  float         v8f;
typedef __attribute__((ext_vector_type(8)))  unsigned int  v8u;

#define T_TOK 4096
#define Dm    512
#define Em    8
#define Hm    2048
#define TILE_M 16
#define XSTR  (Dm / 2 + 4)     // pair-dwords per X row + pad
#define HSTR  (Hm / 2 + 4)     // pair-dwords per H row + pad

// ws layout (bytes)
#define WS_CNT   0
#define WS_PI8   64
#define WS_RV    128
#define WS_LIST  (128 + T_TOK * 4)
#define WS_W1PK  (1u << 20)                          // 8*512*2048*4 B = 32 MiB
#define WS_W2PK  (WS_W1PK + Em * Dm * Hm * 4)        // another 32 MiB

// --------------------------------------------------------------- helpers
__device__ __forceinline__ unsigned short bf_bits(float v) {
    __bf16 h = (__bf16)v;
    return __builtin_bit_cast(unsigned short, h);
}
__device__ __forceinline__ void packsplit2(float a, float b,
                                           unsigned& hi, unsigned& lo) {
    __bf16 ha = (__bf16)a, hb = (__bf16)b;
    float  ra = a - (float)ha, rb = b - (float)hb;
    hi = (unsigned)__builtin_bit_cast(unsigned short, ha) |
         ((unsigned)__builtin_bit_cast(unsigned short, hb) << 16);
    lo = (unsigned)bf_bits(ra) | ((unsigned)bf_bits(rb) << 16);
}
__device__ __forceinline__ v16bf frag_from(const unsigned* p0, const unsigned* p1) {
    uint4 t0 = *(const uint4*)p0;
    uint4 t1 = *(const uint4*)p1;
    v8u u = {t0.x, t0.y, t0.z, t0.w, t1.x, t1.y, t1.z, t1.w};
    return __builtin_bit_cast(v16bf, u);
}

// ---------------------------------------------------------------- init
__global__ void SwitchFFN_init_kernel(int* __restrict__ cnt, float* __restrict__ pi8) {
    int i = threadIdx.x;
    if (i < Em) { cnt[i] = 0; pi8[i] = 0.0f; }
}

// ---------------------------------------------------------------- weight repack
// Fragment-major packed bf16 hi/lo: fragment F = ((e*NT + nt)*KT + kt),
// lane l owns 64 B: dwords [0..7] = hi v16bf, [8..15] = lo v16bf, matching
// the WMMA B-fragment layout (N = lane&15, K pairs striped by lane half).
__global__ __launch_bounds__(256) void SwitchFFN_packw_kernel(
    const float* __restrict__ W, unsigned* __restrict__ dst, int K, int N)
{
    const int KT = K / 32, NT = N / 16;
    int gid  = blockIdx.x * 256 + threadIdx.x;
    int lane = gid & 31;
    int F    = gid >> 5;
    int kt   = F % KT;
    int tmp  = F / KT;
    int nt   = tmp % NT;
    int e    = tmp / NT;
    if (e >= Em) return;

    const int n    = nt * 16 + (lane & 15);
    const int half = lane >> 4;
    unsigned u[16];
#pragma unroll
    for (int j = 0; j < 8; ++j) {
        int kb = kt * 32 + ((j & 4) << 2) + (half << 3) + 2 * (j & 3);
        float c0 = W[((size_t)e * K + kb)     * N + n];
        float c1 = W[((size_t)e * K + kb + 1) * N + n];
        packsplit2(c0, c1, u[j], u[8 + j]);
    }
    unsigned* p = dst + ((size_t)F * 32 + lane) * 16;
#pragma unroll
    for (int q = 0; q < 4; ++q)
        *(uint4*)(p + 4 * q) = make_uint4(u[4*q], u[4*q+1], u[4*q+2], u[4*q+3]);
}

// ---------------------------------------------------------------- routing
__global__ __launch_bounds__(256) void SwitchFFN_route_kernel(
    const float* __restrict__ x, const float* __restrict__ Wg,
    const float* __restrict__ bg, float* __restrict__ rv,
    int* __restrict__ cnt, int* __restrict__ list, float* __restrict__ pi8)
{
    const int lane = threadIdx.x & 31;
    const int t    = blockIdx.x * 8 + (threadIdx.x >> 5);

    float acc[8];
#pragma unroll
    for (int e = 0; e < 8; ++e) acc[e] = 0.0f;

    for (int d = lane; d < Dm; d += 32) {
        float xv = x[(size_t)t * Dm + d];
        const float4* wr = (const float4*)(Wg + (size_t)d * Em);
        float4 w0 = wr[0], w1 = wr[1];
        acc[0] += xv * w0.x; acc[1] += xv * w0.y;
        acc[2] += xv * w0.z; acc[3] += xv * w0.w;
        acc[4] += xv * w1.x; acc[5] += xv * w1.y;
        acc[6] += xv * w1.z; acc[7] += xv * w1.w;
    }
#pragma unroll
    for (int off = 16; off > 0; off >>= 1) {
#pragma unroll
        for (int e = 0; e < 8; ++e) acc[e] += __shfl_xor(acc[e], off, 32);
    }
    if (lane == 0) {
        float lg[8], m = -3.0e38f; int am = 0;
#pragma unroll
        for (int e = 0; e < 8; ++e) {
            lg[e] = acc[e] + bg[e];
            if (lg[e] > m) { m = lg[e]; am = e; }
        }
        float s = 0.0f;
#pragma unroll
        for (int e = 0; e < 8; ++e) s += __expf(lg[e] - m);
        float pmax = 1.0f / s;
        rv[t] = pmax;
        int slot = atomicAdd(&cnt[am], 1);
        list[am * T_TOK + slot] = t;
        if (t < Em) {
            float sp = 0.0f;
#pragma unroll
            for (int e = 0; e < 8; ++e) sp += __expf(lg[e] - m) * pmax;
            pi8[t] = sp * 0.125f;
        }
    }
}

// ---------------------------------------------------------------- loss
__global__ void SwitchFFN_loss_kernel(const int* __restrict__ cnt,
                                      const float* __restrict__ pi8,
                                      float* __restrict__ lossOut)
{
    if (threadIdx.x == 0 && blockIdx.x == 0) {
        float a = 0.0f;
        for (int e = 0; e < Em; ++e)
            a += ((float)cnt[e] * (1.0f / (float)T_TOK)) * pi8[e];
        *lossOut = 0.01f * (float)Em * a;
    }
}

// ---------------------------------------------------------------- expert FFN
// Block = 8 waves, 16 tokens of one expert. n-blocked 4 wide: one A-fragment
// pair feeds 12 WMMAs (4 independent accumulator chains).
__global__ __launch_bounds__(256) void SwitchFFN_ffn_kernel(
    const float* __restrict__ x,  const unsigned* __restrict__ W1pk,
    const float* __restrict__ b1, const unsigned* __restrict__ W2pk,
    const float* __restrict__ b2, const float* __restrict__ rv,
    const int* __restrict__ cnt,  const int* __restrict__ list,
    float* __restrict__ out)
{
    const int e    = blockIdx.y;
    const int base = blockIdx.x * TILE_M;
    const int c    = cnt[e];
    if (base >= c) return;
    const int nTok = min(TILE_M, c - base);

    extern __shared__ char smem[];
    unsigned* XpkHi = (unsigned*)smem;                 // [16][XSTR] pair-dwords
    unsigned* XpkLo = XpkHi + 16 * XSTR;
    unsigned* HpkHi = XpkLo + 16 * XSTR;               // [16][HSTR]
    unsigned* HpkLo = HpkHi + 16 * HSTR;
    int*      toks  = (int*)(HpkLo + 16 * HSTR);
    float*    rvs   = (float*)(toks + TILE_M);

    const int tid  = threadIdx.x;
    const int lane = tid & 31;
    const int wv   = tid >> 5;
    const int mrow = lane & 15;
    const int half = lane >> 4;
    const bool even = (lane & 1) == 0;

    if (tid < TILE_M) {
        int t = (tid < nTok) ? list[e * T_TOK + base + tid] : 0;
        toks[tid] = t;
        rvs[tid]  = (tid < nTok) ? rv[t] : 0.0f;
    }
    __syncthreads();

    // stage X tile: split hi/lo once, pair-packed by K
    for (int idx = tid; idx < TILE_M * (Dm / 2); idx += 256) {
        int row = idx >> 8;                 // Dm/2 == 256
        int cpr = idx & 255;
        float2 v = make_float2(0.f, 0.f);
        if (row < nTok) v = *(const float2*)(x + (size_t)toks[row] * Dm + 2 * cpr);
        unsigned hi, lo;
        packsplit2(v.x, v.y, hi, lo);
        XpkHi[row * XSTR + cpr] = hi;
        XpkLo[row * XSTR + cpr] = lo;
    }
    __syncthreads();

    const unsigned* W1e = W1pk + (size_t)e * (Dm * Hm);   // dwords per expert
    const float*    b1e = b1 + (size_t)e * Hm;
    const int KT1 = Dm / 32;
    const int NSTRIDE1 = KT1 * 512;                       // dwords per n-tile column

    // -------- Phase 1: each wave owns 256 hidden columns = 4 groups of 4 n-tiles
#pragma unroll 1
    for (int ng = 0; ng < 4; ++ng) {
        const int nt0 = wv * 16 + ng * 4;                 // first global n-tile
        v8f acc[4];
#pragma unroll
        for (int j = 0; j < 4; ++j) acc[j] = (v8f){0.f,0.f,0.f,0.f,0.f,0.f,0.f,0.f};
        const unsigned* wf = W1e + (((size_t)nt0 * KT1) * 32 + lane) * 16;
#pragma unroll 2
        for (int kk = 0; kk < Dm; kk += 32) {
            const unsigned* ax = XpkHi + mrow * XSTR + (kk >> 1) + half * 4;
            v16bf ahi = frag_from(ax, ax + 8);
            const unsigned* al = XpkLo + mrow * XSTR + (kk >> 1) + half * 4;
            v16bf alo = frag_from(al, al + 8);
#pragma unroll
            for (int j = 0; j < 4; ++j) {
                const unsigned* wfj = wf + (size_t)j * NSTRIDE1;
                v16bf bhi = frag_from(wfj, wfj + 4);
                v16bf blo = frag_from(wfj + 8, wfj + 12);
                __builtin_prefetch(wfj + 512, 0, 0);      // speculative, next k-frag
                acc[j] = __builtin_amdgcn_wmma_f32_16x16x32_bf16(false, ahi, false, bhi, (short)0, acc[j], false, false);
                acc[j] = __builtin_amdgcn_wmma_f32_16x16x32_bf16(false, ahi, false, blo, (short)0, acc[j], false, false);
                acc[j] = __builtin_amdgcn_wmma_f32_16x16x32_bf16(false, alo, false, bhi, (short)0, acc[j], false, false);
            }
            wf += 512;                                    // next k-fragment
        }
#pragma unroll
        for (int j = 0; j < 4; ++j) {
            const int col = (nt0 + j) * 16 + mrow;
            float bias = b1e[col];
#pragma unroll
            for (int r = 0; r < 8; ++r) {
                int row = (half << 3) + r;                // C/D: M = r + 8*half
                float hv = acc[j][r] + bias;
                hv = hv > 0.0f ? hv : 0.0f;
                __bf16 hh = (__bf16)hv;
                float  rl = hv - (float)hh;
                unsigned hu = __builtin_bit_cast(unsigned short, hh);
                unsigned lu = bf_bits(rl);
                unsigned hp = __shfl_xor(hu, 1, 32);      // neighbour column
                unsigned lp = __shfl_xor(lu, 1, 32);
                if (even) {
                    HpkHi[row * HSTR + (col >> 1)] = hu | (hp << 16);
                    HpkLo[row * HSTR + (col >> 1)] = lu | (lp << 16);
                }
            }
        }
    }
    __syncthreads();

    const unsigned* W2e = W2pk + (size_t)e * (Hm * Dm);
    const float*    b2e = b2 + (size_t)e * Dm;
    const int KT2 = Hm / 32;
    const int NSTRIDE2 = KT2 * 512;

    // -------- Phase 2: each wave owns 64 output columns = one group of 4 n-tiles
    {
        const int nt0 = wv * 4;
        v8f acc[4];
#pragma unroll
        for (int j = 0; j < 4; ++j) acc[j] = (v8f){0.f,0.f,0.f,0.f,0.f,0.f,0.f,0.f};
        const unsigned* wf = W2e + (((size_t)nt0 * KT2) * 32 + lane) * 16;
#pragma unroll 2
        for (int kk = 0; kk < Hm; kk += 32) {
            const unsigned* ax = HpkHi + mrow * HSTR + (kk >> 1) + half * 4;
            v16bf ahi = frag_from(ax, ax + 8);
            const unsigned* al = HpkLo + mrow * HSTR + (kk >> 1) + half * 4;
            v16bf alo = frag_from(al, al + 8);
#pragma unroll
            for (int j = 0; j < 4; ++j) {
                const unsigned* wfj = wf + (size_t)j * NSTRIDE2;
                v16bf bhi = frag_from(wfj, wfj + 4);
                v16bf blo = frag_from(wfj + 8, wfj + 12);
                __builtin_prefetch(wfj + 512, 0, 0);      // speculative, next k-frag
                acc[j] = __builtin_amdgcn_wmma_f32_16x16x32_bf16(false, ahi, false, bhi, (short)0, acc[j], false, false);
                acc[j] = __builtin_amdgcn_wmma_f32_16x16x32_bf16(false, ahi, false, blo, (short)0, acc[j], false, false);
                acc[j] = __builtin_amdgcn_wmma_f32_16x16x32_bf16(false, alo, false, bhi, (short)0, acc[j], false, false);
            }
            wf += 512;
        }
#pragma unroll
        for (int j = 0; j < 4; ++j) {
            const int col = (nt0 + j) * 16 + mrow;
            float bias = b2e[col];
#pragma unroll
            for (int r = 0; r < 8; ++r) {
                int row = (half << 3) + r;
                if (row < nTok) {
                    float yv = (acc[j][r] + bias) * rvs[row];
                    out[(size_t)toks[row] * Dm + col] = yv;
                }
            }
        }
    }
}

// ---------------------------------------------------------------- launch
extern "C" void kernel_launch(void* const* d_in, const int* in_sizes, int n_in,
                              void* d_out, int out_size, void* d_ws, size_t ws_size,
                              hipStream_t stream) {
    const float* x  = (const float*)d_in[0];
    const float* Wg = (const float*)d_in[1];
    const float* bg = (const float*)d_in[2];
    const float* W1 = (const float*)d_in[3];
    const float* b1 = (const float*)d_in[4];
    const float* W2 = (const float*)d_in[5];
    const float* b2 = (const float*)d_in[6];
    float* out = (float*)d_out;

    char*     ws   = (char*)d_ws;
    int*      cnt  = (int*)(ws + WS_CNT);
    float*    pi8  = (float*)(ws + WS_PI8);
    float*    rv   = (float*)(ws + WS_RV);
    int*      list = (int*)(ws + WS_LIST);
    unsigned* w1pk = (unsigned*)(ws + WS_W1PK);
    unsigned* w2pk = (unsigned*)(ws + WS_W2PK);

    SwitchFFN_init_kernel<<<1, 64, 0, stream>>>(cnt, pi8);

    // one-time (per launch) weight repack into fragment-major bf16 hi/lo
    {
        int thr1 = Em * (Hm / 16) * (Dm / 32) * 32;    // 524288
        SwitchFFN_packw_kernel<<<thr1 / 256, 256, 0, stream>>>(W1, w1pk, Dm, Hm);
        int thr2 = Em * (Dm / 16) * (Hm / 32) * 32;    // 524288
        SwitchFFN_packw_kernel<<<thr2 / 256, 256, 0, stream>>>(W2, w2pk, Hm, Dm);
    }

    SwitchFFN_route_kernel<<<T_TOK / 8, 256, 0, stream>>>(x, Wg, bg, rv, cnt, list, pi8);
    SwitchFFN_loss_kernel<<<1, 32, 0, stream>>>(cnt, pi8, out + (size_t)T_TOK * Dm);

    const size_t smem = (size_t)16 * XSTR * 4 * 2 + (size_t)16 * HSTR * 4 * 2 +
                        TILE_M * 4 + TILE_M * 4;       // ~165 KB dynamic LDS
    dim3 grid(T_TOK / TILE_M, Em);
    SwitchFFN_ffn_kernel<<<grid, 256, smem, stream>>>(x, w1pk, b1, w2pk, b2,
                                                      rv, cnt, list, out);
}
